// M2BertAttention_81389630259284
// MI455X (gfx1250) — matrix operand order repaired and checked by reading, not compiled, not run on hardware
//
#include <hip/hip_runtime.h>

// ---------------- problem constants ----------------
constexpr int Hdim = 768;
constexpr int NBv  = 16;
constexpr int BSZv = 48;
constexpr int Ttot = 4 * 4096;     // 16384 tokens
constexpr int TM   = 64;           // tokens per workgroup
constexpr int KSTEP = 32;          // WMMA bf16 K per step
constexpr int NKB  = Hdim / KSTEP; // 24
constexpr int NSTG = 4 * NKB;      // 96 stages per phase (4 col chunks x 24 K steps)
constexpr int XROW = 776;          // padded LDS row stride (bf16 elems) for 768-wide rows
constexpr int PROW = 40;           // padded LDS panel row stride (bf16 elems) for 32-wide rows
constexpr int YROW = 768;          // f32 y tile stride (unpadded; only scattered b32 access)

// ---------------- LDS layout (bytes) ----------------
constexpr int OFF_HA  = 0;         // 64*776*2 = 99328  h tile (bf16)    — live both phases
constexpr int OFF_XA  = 99328;     // 99328             x tile (bf16)    — phase 1
constexpr int OFF_PM0 = 198656;    // 192*40*2 = 15360  monarch panel 0  — phase 1
constexpr int OFF_PG0 = 214016;    // 15360             gate panel 0     — phase 1
constexpr int OFF_PM1 = 229376;    // 15360             monarch panel 1  — phase 1
constexpr int OFF_PG1 = 244736;    // 15360 -> 260096   gate panel 1     — phase 1
constexpr int OFF_Y   = 99328;     // 64*768*4 = 196608 -> 295936  f32 y — phase 2 (overlaps XA+P*)
constexpr int OFF_PO0 = 295936;    // 15360             out panel 0      — phase 2
constexpr int OFF_PO1 = 311296;    // 15360 -> 326656   out panel 1      — phase 2
constexpr int SMEM_BYTES = 326656; // <= 320 KB per WGP

typedef __attribute__((ext_vector_type(16))) __bf16 v16bf;
typedef __attribute__((ext_vector_type(8)))  float  v8f;

union Frag16 { uint4 u[2]; v16bf v; };

__device__ __forceinline__ unsigned short f2bf(float f) {
  unsigned int u = __float_as_uint(f);
  u += 0x7FFFu + ((u >> 16) & 1u);            // round-to-nearest-even
  return (unsigned short)(u >> 16);
}

// ---- CDNA5 async Global->LDS copy (bypasses VGPRs, tracked by ASYNCcnt) ----
__device__ __forceinline__ void async_b128(unsigned ldsByteAddr, const void* gaddr) {
  asm volatile("global_load_async_to_lds_b128 %0, %1, off"
               :: "v"(ldsByteAddr), "v"(gaddr) : "memory");
}
__device__ __forceinline__ unsigned lds_off(const void* p) {
  return (unsigned)(uintptr_t)p;              // flat->LDS: low 32 bits are the LDS byte offset
}
#define WAIT_ASYNC(n) asm volatile("s_wait_asynccnt " #n ::: "memory")

// ---------------------------------------------------------------------------
// Kernel 0: materialize dense Monarch matrix M (bf16) and convert gate/out
// weights to bf16.  M[o=p*48+q][in=k*48+i] = w1[k,i,q] * w2[q,k,p].
// ---------------------------------------------------------------------------
__global__ void prep_weights(const float* __restrict__ w1,
                             const float* __restrict__ w2,
                             const float* __restrict__ gate_w,
                             const float* __restrict__ out_w,
                             unsigned short* __restrict__ Mw,
                             unsigned short* __restrict__ Gw,
                             unsigned short* __restrict__ Ow)
{
  int idx = blockIdx.x * 256 + threadIdx.x;
  if (idx >= Hdim * Hdim) return;
  int o  = idx / Hdim;
  int in = idx - o * Hdim;
  int p = o / BSZv, q = o - p * BSZv;
  int k = in / BSZv, i = in - k * BSZv;
  float m = w1[(k * BSZv + i) * BSZv + q] * w2[(q * NBv + k) * NBv + p];
  Mw[idx] = f2bf(m);
  Gw[idx] = f2bf(gate_w[idx]);
  Ow[idx] = f2bf(out_w[idx]);
}

// ---------------------------------------------------------------------------
// Kernel 1: fused  mixed = x@M^T ; gate = sigmoid(x@Gw^T+gb) ; h = mixed*gate ;
//                  y = h@Ow^T + ob + x ; out = LayerNorm(y)
// 64 tokens / workgroup, 8 wave32 waves, bf16 WMMA + f32 accum,
// double-buffered async Global->LDS weight-panel pipeline.
// ---------------------------------------------------------------------------
__global__ void __launch_bounds__(256, 1)
m2bert_fused(const float* __restrict__ x,
             const unsigned short* __restrict__ Mw,
             const unsigned short* __restrict__ Gw,
             const unsigned short* __restrict__ Ow,
             const float* __restrict__ gate_b,
             const float* __restrict__ out_b,
             const float* __restrict__ gamma,
             const float* __restrict__ beta,
             float* __restrict__ out)
{
  extern __shared__ char smem[];
  unsigned short* HA = (unsigned short*)(smem + OFF_HA);
  unsigned short* XA = (unsigned short*)(smem + OFF_XA);
  float*          Y  = (float*)(smem + OFF_Y);

  // buffer selectors computed at point of use — NO pointer aggregates from
  // addrspace(3) (lld cannot relocate addrspacecast in static initializers)
  auto PMp = [&](int b) { return (unsigned short*)(smem + (b ? OFF_PM1 : OFF_PM0)); };
  auto PGp = [&](int b) { return (unsigned short*)(smem + (b ? OFF_PG1 : OFF_PG0)); };
  auto POp = [&](int b) { return (unsigned short*)(smem + (b ? OFF_PO1 : OFF_PO0)); };

  const int tid  = threadIdx.x;
  const int lane = tid & 31;
  const int wid  = tid >> 5;
  const int n16  = lane & 15;
  const int loHalf = (lane < 16) ? 1 : 0;
  const int khalfA = loHalf ? 0 : 16;   // A frag: lanes 0-15 -> K0..15,       16-31 -> K16..31
  const int khalfB = loHalf ? 0 : 16;   // B frag: lanes 0-15 -> K0..15,       16-31 -> K16..31
  const int rowTile = wid & 3;          // 4 row tiles of 16 tokens
  const int colHalf = wid >> 2;         // 2 column halves of 384
  const int tokBase = blockIdx.x * TM;

  v8f zero8;
  #pragma unroll
  for (int j = 0; j < 8; ++j) zero8[j] = 0.0f;

  // stage s -> (cc = s/NKB column chunk, kb = s%NKB K step); panel row -> matrix row
  auto stage2 = [&](const unsigned short* W0, const unsigned short* W1,
                    unsigned short* L0, unsigned short* L1, int s) {
    int cc = s / NKB, kb = s - cc * NKB;
    #pragma unroll
    for (int i = 0; i < 3; ++i) {
      int seg  = tid + i * 256;         // 768 16B segments per matrix
      int prow = seg >> 2;              // 0..191
      int part = seg & 3;
      int o = (prow < 96) ? (cc * 96 + prow) : (384 + cc * 96 + (prow - 96));
      int goff = o * Hdim + kb * KSTEP + part * 8;
      int loff = prow * PROW + part * 8;
      async_b128(lds_off(L0 + loff), W0 + goff);          // 6 async ops / wave / stage
      async_b128(lds_off(L1 + loff), W1 + goff);
    }
  };
  auto stage1 = [&](const unsigned short* W, unsigned short* L, int s) {
    int cc = s / NKB, kb = s - cc * NKB;
    #pragma unroll
    for (int i = 0; i < 3; ++i) {
      int seg  = tid + i * 256;
      int prow = seg >> 2;
      int part = seg & 3;
      int o = (prow < 96) ? (cc * 96 + prow) : (384 + cc * 96 + (prow - 96));
      async_b128(lds_off(L + prow * PROW + part * 8),
                 W + o * Hdim + kb * KSTEP + part * 8);   // 3 async ops / wave / stage
    }
  };

  // kick off stage 0 of phase 1 so the DMA overlaps the x-tile conversion
  stage2(Mw, Gw, PMp(0), PGp(0), 0);

  // ---- stage x tile -> LDS bf16 (row stride 776 kills bank conflicts) ----
  {
    const float4* xg = (const float4*)(x + (size_t)tokBase * Hdim);
    #pragma unroll
    for (int i = 0; i < 48; ++i) {
      int seg = tid + i * 256;          // 12288 float4 segments
      float4 v = xg[seg];
      int e = seg * 4;
      int r = e / Hdim;
      int c = e - r * Hdim;
      ushort4 us;
      us.x = f2bf(v.x); us.y = f2bf(v.y); us.z = f2bf(v.z); us.w = f2bf(v.w);
      *(ushort4*)(XA + r * XROW + c) = us;
    }
  }

  // =======================  Phase 1: mixed & gate  =======================
  for (int cc = 0; cc < 4; ++cc) {
    const int colW = colHalf * 384 + cc * 96;   // this wave's 96 columns
    v8f accM[6], accG[6];
    #pragma unroll
    for (int t = 0; t < 6; ++t) { accM[t] = zero8; accG[t] = zero8; }

    for (int kb = 0; kb < NKB; ++kb) {
      const int s   = cc * NKB + kb;
      const int buf = s & 1;
      if (s + 1 < NSTG) {               // prefetch next panel pair (fire & forget)
        stage2(Mw, Gw, PMp((s + 1) & 1), PGp((s + 1) & 1), s + 1);
        WAIT_ASYNC(6);                  // in-order completion: stage s is done
      } else {
        WAIT_ASYNC(0);
      }
      __syncthreads();                  // everyone's stage s landed in LDS

      Frag16 a;
      {
        const unsigned short* ap =
            XA + (rowTile * 16 + n16) * XROW + kb * KSTEP + khalfA;
        a.u[0] = *(const uint4*)(ap);
        a.u[1] = *(const uint4*)(ap + 8);
      }
      const unsigned short* PM = PMp(buf);
      const unsigned short* PG = PGp(buf);
      #pragma unroll
      for (int t = 0; t < 6; ++t) {
        Frag16 bm, bg;
        const unsigned short* bp = PM + (colHalf * 96 + t * 16 + n16) * PROW + khalfB;
        const unsigned short* gp = PG + (colHalf * 96 + t * 16 + n16) * PROW + khalfB;
        bm.u[0] = *(const uint4*)(bp); bm.u[1] = *(const uint4*)(bp + 8);
        bg.u[0] = *(const uint4*)(gp); bg.u[1] = *(const uint4*)(gp + 8);
        accM[t] = __builtin_amdgcn_wmma_f32_16x16x32_bf16(
            false, a.v, false, bm.v, (short)0, accM[t], false, false);
        accG[t] = __builtin_amdgcn_wmma_f32_16x16x32_bf16(
            false, a.v, false, bg.v, (short)0, accG[t], false, false);
      }
      __syncthreads();                  // all reads of buf done before stage s+2 rewrites it
    }

    // epilogue: h = mixed * sigmoid(gate + gate_b)  -> HA (bf16)
    #pragma unroll
    for (int t = 0; t < 6; ++t) {
      int col = colW + t * 16 + n16;
      float gb = gate_b[col];
      #pragma unroll
      for (int j = 0; j < 8; ++j) {
        int row = rowTile * 16 + j + (loHalf ? 0 : 8);
        float g = 1.0f / (1.0f + __expf(-(accG[t][j] + gb)));
        HA[row * XROW + col] = f2bf(accM[t][j] * g);
      }
    }
  }
  __syncthreads();

  // =======================  Phase 2: output GEMM  =======================
  stage1(Ow, POp(0), 0);
  for (int cc = 0; cc < 4; ++cc) {
    const int colW = colHalf * 384 + cc * 96;
    v8f accO[6];
    #pragma unroll
    for (int t = 0; t < 6; ++t) accO[t] = zero8;

    for (int kb = 0; kb < NKB; ++kb) {
      const int s   = cc * NKB + kb;
      const int buf = s & 1;
      if (s + 1 < NSTG) {
        stage1(Ow, POp((s + 1) & 1), s + 1);
        WAIT_ASYNC(3);
      } else {
        WAIT_ASYNC(0);
      }
      __syncthreads();

      Frag16 a;
      {
        const unsigned short* ap =
            HA + (rowTile * 16 + n16) * XROW + kb * KSTEP + khalfA;
        a.u[0] = *(const uint4*)(ap);
        a.u[1] = *(const uint4*)(ap + 8);
      }
      const unsigned short* PO = POp(buf);
      #pragma unroll
      for (int t = 0; t < 6; ++t) {
        Frag16 bo;
        const unsigned short* bp = PO + (colHalf * 96 + t * 16 + n16) * PROW + khalfB;
        bo.u[0] = *(const uint4*)(bp); bo.u[1] = *(const uint4*)(bp + 8);
        accO[t] = __builtin_amdgcn_wmma_f32_16x16x32_bf16(
            false, a.v, false, bo.v, (short)0, accO[t], false, false);
      }
      __syncthreads();
    }

    // epilogue: y = h@Ow^T + out_b + residual -> Y (f32 LDS)
    #pragma unroll
    for (int t = 0; t < 6; ++t) {
      int col = colW + t * 16 + n16;
      float ob = out_b[col];
      #pragma unroll
      for (int j = 0; j < 8; ++j) {
        int row = rowTile * 16 + j + (loHalf ? 0 : 8);
        float resid = x[(size_t)(tokBase + row) * Hdim + col];
        Y[row * YROW + col] = accO[t][j] + ob + resid;
      }
    }
  }
  __syncthreads();

  // =======================  LayerNorm  =======================
  // each wave normalizes 8 rows; wave32 shuffle reduction over 768 cols
  #pragma unroll
  for (int rr = 0; rr < 8; ++rr) {
    int row = wid * 8 + rr;
    float s = 0.0f, s2 = 0.0f;
    for (int c = lane; c < Hdim; c += 32) {
      float v = Y[row * YROW + c];
      s += v; s2 += v * v;
    }
    #pragma unroll
    for (int m = 16; m >= 1; m >>= 1) {
      s  += __shfl_xor(s,  m, 32);
      s2 += __shfl_xor(s2, m, 32);
    }
    const float invH = 1.0f / (float)Hdim;
    float mu  = s * invH;
    float var = s2 * invH - mu * mu;
    float inv = rsqrtf(var + 1e-12f);
    float* op = out + (size_t)(tokBase + row) * Hdim;
    for (int c = lane; c < Hdim; c += 32) {
      float v = Y[row * YROW + c];
      op[c] = (v - mu) * inv * gamma[c] + beta[c];
    }
  }
}

// ---------------------------------------------------------------------------
extern "C" void kernel_launch(void* const* d_in, const int* in_sizes, int n_in,
                              void* d_out, int out_size, void* d_ws, size_t ws_size,
                              hipStream_t stream) {
  const float* x      = (const float*)d_in[0];
  const float* w1     = (const float*)d_in[1];
  const float* w2     = (const float*)d_in[2];
  const float* gate_w = (const float*)d_in[3];
  const float* gate_b = (const float*)d_in[4];
  const float* out_w  = (const float*)d_in[5];
  const float* out_b  = (const float*)d_in[6];
  const float* gamma  = (const float*)d_in[7];
  const float* beta   = (const float*)d_in[8];
  float* out = (float*)d_out;

  unsigned short* Mw = (unsigned short*)d_ws;        // 768*768 bf16 monarch matrix
  unsigned short* Gw = Mw + Hdim * Hdim;             // gate weights bf16
  unsigned short* Ow = Gw + Hdim * Hdim;             // out weights bf16

  prep_weights<<<(Hdim * Hdim + 255) / 256, 256, 0, stream>>>(
      w1, w2, gate_w, out_w, Mw, Gw, Ow);

  (void)hipFuncSetAttribute(reinterpret_cast<const void*>(m2bert_fused),
                            hipFuncAttributeMaxDynamicSharedMemorySize, SMEM_BYTES);
  m2bert_fused<<<Ttot / TM, 256, SMEM_BYTES, stream>>>(
      x, Mw, Gw, Ow, gate_b, out_b, gamma, beta, out);
}
